// DenseEdgeConv_25151328485702
// MI455X (gfx1250) — compile-verified, matrix operand order checked by reading, not run on hardware
//
#include <hip/hip_runtime.h>
#include <hip/hip_bf16.h>
#include <math.h>

typedef __attribute__((ext_vector_type(16))) _Float16 v16h;
typedef __attribute__((ext_vector_type(8)))  float    v8f;
typedef __attribute__((ext_vector_type(4)))  int      v4i;

#define BB 8
#define NN 4096
#define DD 64
#define GG 32
#define KK 16
#define PPW 2   // points per wave: every B fragment / Wg row serves PPW WMMAs

#if defined(__has_builtin)
#if __has_builtin(__builtin_amdgcn_global_load_async_to_lds_b128)
#define HAVE_ASYNC_LDS 1
#endif
#endif

// Typed address-space pointers for the gfx1250 async global->LDS copy builtin.
typedef __attribute__((address_space(1))) v4i* gv4i_p;
typedef __attribute__((address_space(3))) v4i* lv4i_p;

// Wave-local LDS fence: lanes of a wave32 run in lockstep, so after the
// wave's own ds_stores complete (DScnt==0) its subsequent ds_loads see them.
__device__ __forceinline__ void lds_fence() {
    asm volatile("s_wait_dscnt 0x0" ::: "memory");
}
__device__ __forceinline__ void async_fence() {
    asm volatile("s_wait_asynccnt 0x0" ::: "memory");
}

// ---------------------------------------------------------------------------
// Pack an fp32 row-major weight matrix W[k][n] (Kdim x Ndim) into f16 WMMA
// B-fragments matching the CDNA5 16-bit operand layout:
//   lane l: N = nt*16 + (l&15); kbase = (l<16)?0:8
//   VGPR v<4: K = kbase+2v{+1};  v>=4: K = 16+kbase+2(v-4){+1}
// Fragment f = kt*(Ndim/16)+nt stored as 32 lanes x 16 halves contiguously,
// so the GEMM loop fetches one v16h (32B) per lane per fragment.
// ---------------------------------------------------------------------------
__global__ void pack_wmma_b(const float* __restrict__ W, _Float16* __restrict__ dst,
                            int Kdim, int Ndim) {
    int tid = blockIdx.x * 256 + threadIdx.x;
    int total = Kdim * Ndim;
    if (tid >= total) return;
    int numNT = Ndim >> 4;
    int h    = tid & 15;
    int lane = (tid >> 4) & 31;
    int f    = tid >> 9;
    int nt = f % numNT;
    int kt = f / numNT;
    int n = nt * 16 + (lane & 15);
    int kbase = (lane < 16) ? 0 : 8;
    int v = h >> 1, p = h & 1;
    int kk = (v < 4) ? (kbase + 2 * v + p) : (16 + kbase + 2 * (v - 4) + p);
    int k = kt * 32 + kk;
    dst[tid] = (_Float16)W[k * Ndim + n];
}

// ---------------------------------------------------------------------------
// KNN: per-thread query, candidates tiled through LDS, register top-17 list
// (fully unrolled insertion -> no scratch). Self (d==0) lands in slot 0 and
// is dropped, matching the reference's offset=1.
// ---------------------------------------------------------------------------
__global__ void __launch_bounds__(256)
knn_kernel(const float* __restrict__ pos, int* __restrict__ idx_out) {
    __shared__ float cx[256], cy[256], cz[256];
    const int tid = threadIdx.x;
    const int b = blockIdx.x >> 4;                 // 16 blocks per batch
    const int q = ((blockIdx.x & 15) << 8) + tid;
    const float* pb = pos + (size_t)b * NN * 3;
    const float qx = pb[q * 3 + 0], qy = pb[q * 3 + 1], qz = pb[q * 3 + 2];

    float bd[KK + 1]; int bi[KK + 1];
#pragma unroll
    for (int i = 0; i <= KK; ++i) { bd[i] = 3.4e38f; bi[i] = 0; }

    for (int tile = 0; tile < NN / 256; ++tile) {
        int c = tile * 256 + tid;
        cx[tid] = pb[c * 3 + 0];
        cy[tid] = pb[c * 3 + 1];
        cz[tid] = pb[c * 3 + 2];
        __syncthreads();
        for (int j = 0; j < 256; ++j) {
            float dx = qx - cx[j], dy = qy - cy[j], dz = qz - cz[j];
            float d = dx * dx + dy * dy + dz * dz;
            if (d < bd[KK]) {                      // threshold gate: rare path
                float cd = d; int ci = tile * 256 + j;
#pragma unroll
                for (int p = 0; p <= KK; ++p) {    // unrolled sorted insert
                    if (cd < bd[p]) {
                        float td = bd[p]; int ti = bi[p];
                        bd[p] = cd; bi[p] = ci;
                        cd = td; ci = ti;
                    }
                }
            }
        }
        __syncthreads();
    }
    int* op = idx_out + (size_t)(b * NN + q) * KK;
#pragma unroll
    for (int k = 0; k < KK; ++k) op[k] = bi[k + 1];
}

// Per-point LDS workspace. Strides chosen for conflict-free per-row access:
//   nbr stride 68 floats  -> dword bank advances by 4 per row
//   hbuf stride 264 halves-> dword bank advances by 4 per row
//   y2 stride 136 halves  -> dword bank advances by 4 per row
struct PointLds {
    float    xc[DD];           // center features
    int      nidx[KK];         // neighbor ids
    float    nbr[KK * 68];     // neighbor features (padded)
    _Float16 hbuf[16 * 264];   // h after layer_first.1 (16 x 256)
    _Float16 y2[16 * 136];     // [mid | h2 | x] (16 x 128)
    float    meanb[128];
    float    gateb[128];
};

// Build a 16-bit WMMA A-fragment from an LDS f16 matrix (row stride, chan off).
__device__ __forceinline__ v16h build_a_lds(const _Float16* base, int stride,
                                            int coff, int lane) {
    const int row = lane & 15;
    const int kbase = (lane < 16) ? 0 : 8;
    v16h a;
#pragma unroll
    for (int v = 0; v < 8; ++v) {
        int kk0 = (v < 4) ? (kbase + 2 * v) : (16 + kbase + 2 * (v - 4));
        a[2 * v]     = base[row * stride + coff + kk0];
        a[2 * v + 1] = base[row * stride + coff + kk0 + 1];
    }
    return a;
}

// ---------------------------------------------------------------------------
// Fused DenseEdgeConv: one wave32 per PPW points (M=16 neighbors = one WMMA
// tile). Every B fragment load and every Wg row feeds PPW accumulators.
// ---------------------------------------------------------------------------
__global__ void __launch_bounds__(32)
edgeconv_kernel(const float* __restrict__ x, const int* __restrict__ knn,
                const _Float16* __restrict__ w1p, const _Float16* __restrict__ w2p,
                const _Float16* __restrict__ wmidp, const _Float16* __restrict__ wlastp,
                const float* __restrict__ b1, const float* __restrict__ b2,
                const float* __restrict__ bmid, const float* __restrict__ Wg,
                const float* __restrict__ bg, const float* __restrict__ blast,
                float* __restrict__ out) {
    __shared__ PointLds S[PPW];
    const int lane = threadIdx.x & 31;
    const int pbase = blockIdx.x * PPW;
    const int lane15 = lane & 15;
    const int kbase = (lane < 16) ? 0 : 8;
    const int rbase = (lane < 16) ? 0 : 8;

    // ---- stage center features + neighbor ids for both points ----
#pragma unroll
    for (int p = 0; p < PPW; ++p) {
        const int point = pbase + p;
        const float* xrow = x + (size_t)point * DD;
        S[p].xc[lane] = xrow[lane];
        S[p].xc[lane + 32] = xrow[lane + 32];
        if (lane < KK) S[p].nidx[lane] = knn[(size_t)point * KK + lane];
    }
    lds_fence();

    // ---- stage 16 neighbor rows per point (global -> LDS) ----
#pragma unroll
    for (int p = 0; p < PPW; ++p) {
        const int bb = (pbase + p) >> 12;             // / NN
#pragma unroll
        for (int i = 0; i < 8; ++i) {
            int t = lane + 32 * i;                    // 0..255 float4 slots
            int row = t >> 4, c4 = t & 15;
            const float* g = x + ((size_t)bb * NN + S[p].nidx[row]) * DD + c4 * 4;
#ifdef HAVE_ASYNC_LDS
            // CDNA5 async copy: global -> LDS without a VGPR round trip,
            // tracked by ASYNCcnt.
            __builtin_amdgcn_global_load_async_to_lds_b128(
                (gv4i_p)(void*)g,
                (lv4i_p)&S[p].nbr[row * 68 + c4 * 4],
                0, 0);
#else
            const float4 v = *(const float4*)g;
            *(float4*)&S[p].nbr[row * 68 + c4 * 4] = v;
#endif
        }
    }
#ifdef HAVE_ASYNC_LDS
    async_fence();
#else
    lds_fence();
#endif

    // ---- build A-fragments for edge = [x_t | knn | knn - x_t] (16x192) ----
    v16h a1[PPW][6];
#pragma unroll
    for (int p = 0; p < PPW; ++p) {
#pragma unroll
        for (int kt = 0; kt < 6; ++kt) {
#pragma unroll
            for (int v = 0; v < 8; ++v) {
                int kk0 = (v < 4) ? (kbase + 2 * v) : (16 + kbase + 2 * (v - 4));
#pragma unroll
                for (int q = 0; q < 2; ++q) {
                    int c = kt * 32 + kk0 + q;
                    float e;
                    if (kt < 2)      e = S[p].xc[c];
                    else if (kt < 4) e = S[p].nbr[lane15 * 68 + (c - 64)];
                    else             e = S[p].nbr[lane15 * 68 + (c - 128)] - S[p].xc[c - 128];
                    a1[p][kt][2 * v + q] = (_Float16)e;
                }
            }
        }
    }

    // ---- GEMM1: edge(16x192) @ W1(192x256), relu -> hbuf f16 ----
    // Each B fragment (1KB across the wave) feeds PPW WMMAs. Unroll capped at
    // 4 so live accumulators (4*PPW tiles = 64 VGPRs) + 12 A-fragments stay
    // under 256 architectural VGPRs: avoids s_set_vgpr_msb juggling in the
    // hot loop and keeps wave residency for L2-latency hiding.
#pragma unroll 4
    for (int nt = 0; nt < 16; ++nt) {
        v8f acc[PPW];
#pragma unroll
        for (int p = 0; p < PPW; ++p) acc[p] = (v8f){};
#pragma unroll
        for (int kt = 0; kt < 6; ++kt) {
            v16h bf = *(const v16h*)(w1p + (size_t)((kt * 16 + nt) * 32 + lane) * 16);
#pragma unroll
            for (int p = 0; p < PPW; ++p)
                acc[p] = __builtin_amdgcn_wmma_f32_16x16x32_f16(false, a1[p][kt], false, bf,
                                                                (short)0, acc[p], false, false);
        }
        int col = nt * 16 + lane15;
        float bias = b1[col];
#pragma unroll
        for (int p = 0; p < PPW; ++p)
#pragma unroll
            for (int r = 0; r < 8; ++r)
                S[p].hbuf[(rbase + r) * 264 + col] = (_Float16)fmaxf(acc[p][r] + bias, 0.f);
    }
    lds_fence();

    // ---- GEMM2: h(16x256) @ W2(256x32), relu -> y2 cols 32..63 ----
#pragma unroll
    for (int nt = 0; nt < 2; ++nt) {
        v8f acc[PPW];
#pragma unroll
        for (int p = 0; p < PPW; ++p) acc[p] = (v8f){};
#pragma unroll
        for (int kt = 0; kt < 8; ++kt) {
            v16h bf = *(const v16h*)(w2p + (size_t)((kt * 2 + nt) * 32 + lane) * 16);
#pragma unroll
            for (int p = 0; p < PPW; ++p) {
                v16h a = build_a_lds(S[p].hbuf, 264, kt * 32, lane);
                acc[p] = __builtin_amdgcn_wmma_f32_16x16x32_f16(false, a, false, bf,
                                                                (short)0, acc[p], false, false);
            }
        }
        int col = nt * 16 + lane15;
        float bias = b2[col];
#pragma unroll
        for (int p = 0; p < PPW; ++p)
#pragma unroll
            for (int r = 0; r < 8; ++r)
                S[p].y2[(rbase + r) * 136 + 32 + col] = (_Float16)fmaxf(acc[p][r] + bias, 0.f);
    }
    // x_t broadcast into y2 cols 64..127
#pragma unroll
    for (int p = 0; p < PPW; ++p)
#pragma unroll
        for (int i = 0; i < 32; ++i) {
            int t = lane + 32 * i;                    // 0..1023
            int row = t >> 6, c = t & 63;
            S[p].y2[row * 136 + 64 + c] = (_Float16)S[p].xc[c];
        }
    lds_fence();

    // ---- GEMM-mid: y(16x96) @ Wmid(96x32), relu -> y2 cols 0..31 ----
#pragma unroll
    for (int nt = 0; nt < 2; ++nt) {
        v8f acc[PPW];
#pragma unroll
        for (int p = 0; p < PPW; ++p) acc[p] = (v8f){};
#pragma unroll
        for (int kt = 0; kt < 3; ++kt) {
            v16h bf = *(const v16h*)(wmidp + (size_t)((kt * 2 + nt) * 32 + lane) * 16);
#pragma unroll
            for (int p = 0; p < PPW; ++p) {
                v16h a = build_a_lds(S[p].y2, 136, 32 + kt * 32, lane);
                acc[p] = __builtin_amdgcn_wmma_f32_16x16x32_f16(false, a, false, bf,
                                                                (short)0, acc[p], false, false);
            }
        }
        int col = nt * 16 + lane15;
        float bias = bmid[col];
#pragma unroll
        for (int p = 0; p < PPW; ++p)
#pragma unroll
            for (int r = 0; r < 8; ++r)
                S[p].y2[(rbase + r) * 136 + col] = (_Float16)fmaxf(acc[p][r] + bias, 0.f);
    }
    lds_fence();

    // ---- channel gate: mean over K, 1x128 @ 128x128, sigmoid ----
#pragma unroll
    for (int p = 0; p < PPW; ++p)
#pragma unroll
        for (int q = 0; q < 4; ++q) {
            int ch = lane * 4 + q;
            float s = 0.f;
#pragma unroll
            for (int r = 0; r < 16; ++r) s += (float)S[p].y2[r * 136 + ch];
            S[p].meanb[ch] = s * (1.f / 16.f);
        }
    lds_fence();
    float ga[PPW][4];
#pragma unroll
    for (int p = 0; p < PPW; ++p)
#pragma unroll
        for (int q = 0; q < 4; ++q) ga[p][q] = bg[lane * 4 + q];
    for (int c = 0; c < 128; ++c) {
        const float4 wr = *(const float4*)(Wg + c * 128 + lane * 4);  // shared row
        float w4[4] = {wr.x, wr.y, wr.z, wr.w};
#pragma unroll
        for (int p = 0; p < PPW; ++p) {
            float m = S[p].meanb[c];                  // broadcast read
#pragma unroll
            for (int q = 0; q < 4; ++q) ga[p][q] += m * w4[q];
        }
    }
#pragma unroll
    for (int p = 0; p < PPW; ++p)
#pragma unroll
        for (int q = 0; q < 4; ++q)
            S[p].gateb[lane * 4 + q] = 1.f / (1.f + __expf(-ga[p][q]));
    lds_fence();

    // ---- GEMM-last: (y2*gate)(16x128) @ Wlast(128x32); max over K ----
#pragma unroll
    for (int nt = 0; nt < 2; ++nt) {
        v8f acc[PPW];
#pragma unroll
        for (int p = 0; p < PPW; ++p) acc[p] = (v8f){};
#pragma unroll
        for (int kt = 0; kt < 4; ++kt) {
            v16h bf = *(const v16h*)(wlastp + (size_t)((kt * 2 + nt) * 32 + lane) * 16);
#pragma unroll
            for (int p = 0; p < PPW; ++p) {
                v16h a;
#pragma unroll
                for (int v = 0; v < 8; ++v) {
                    int kk0 = (v < 4) ? (kbase + 2 * v) : (16 + kbase + 2 * (v - 4));
#pragma unroll
                    for (int q = 0; q < 2; ++q) {
                        int c = kt * 32 + kk0 + q;
                        a[2 * v + q] = (_Float16)((float)S[p].y2[lane15 * 136 + c] *
                                                  S[p].gateb[c]);
                    }
                }
                acc[p] = __builtin_amdgcn_wmma_f32_16x16x32_f16(false, a, false, bf,
                                                                (short)0, acc[p], false, false);
            }
        }
        int col = nt * 16 + lane15;
        float bias = blast[col];
#pragma unroll
        for (int p = 0; p < PPW; ++p) {
            float mx = -3.4e38f;
#pragma unroll
            for (int r = 0; r < 8; ++r) mx = fmaxf(mx, acc[p][r] + bias);
            mx = fmaxf(mx, __shfl_xor(mx, 16, 32));   // rows 0-7 vs 8-15
            if (lane < 16) out[(size_t)(pbase + p) * 160 + col] = mx;
        }
    }

    // ---- max over K of gated y2 -> out channels 32..159 ----
#pragma unroll
    for (int p = 0; p < PPW; ++p)
#pragma unroll
        for (int q = 0; q < 4; ++q) {
            int ch = lane * 4 + q;
            float g = S[p].gateb[ch];
            float mx = -3.4e38f;
#pragma unroll
            for (int r = 0; r < 16; ++r)
                mx = fmaxf(mx, (float)S[p].y2[r * 136 + ch] * g);
            out[(size_t)(pbase + p) * 160 + 32 + ch] = mx;
        }
}

extern "C" void kernel_launch(void* const* d_in, const int* in_sizes, int n_in,
                              void* d_out, int out_size, void* d_ws, size_t ws_size,
                              hipStream_t stream) {
    const float* x     = (const float*)d_in[0];
    const float* pos   = (const float*)d_in[1];
    const float* W1    = (const float*)d_in[2];
    const float* b1    = (const float*)d_in[3];
    const float* W2    = (const float*)d_in[4];
    const float* b2    = (const float*)d_in[5];
    const float* Wmid  = (const float*)d_in[6];
    const float* bmid  = (const float*)d_in[7];
    const float* Wg    = (const float*)d_in[8];
    const float* bg    = (const float*)d_in[9];
    const float* Wlast = (const float*)d_in[10];
    const float* blast = (const float*)d_in[11];
    float* out = (float*)d_out;

    char* ws = (char*)d_ws;
    int* d_idx = (int*)ws;
    size_t off = (size_t)BB * NN * KK * sizeof(int);      // 2 MB, 32B-aligned
    _Float16* w1p    = (_Float16*)(ws + off); off += 192 * 256 * 2;
    _Float16* w2p    = (_Float16*)(ws + off); off += 256 * 32 * 2;
    _Float16* wmidp  = (_Float16*)(ws + off); off += 96 * 32 * 2;
    _Float16* wlastp = (_Float16*)(ws + off); off += 128 * 32 * 2;

    pack_wmma_b<<<(192 * 256 + 255) / 256, 256, 0, stream>>>(W1, w1p, 192, 256);
    pack_wmma_b<<<(256 * 32 + 255) / 256, 256, 0, stream>>>(W2, w2p, 256, 32);
    pack_wmma_b<<<(96 * 32 + 255) / 256, 256, 0, stream>>>(Wmid, wmidp, 96, 32);
    pack_wmma_b<<<(128 * 32 + 255) / 256, 256, 0, stream>>>(Wlast, wlastp, 128, 32);

    knn_kernel<<<BB * (NN / 256), 256, 0, stream>>>(pos, d_idx);

    edgeconv_kernel<<<(BB * NN) / PPW, 32, 0, stream>>>(
        x, d_idx, w1p, w2p, wmidp, wlastp, b1, b2, bmid, Wg, bg, blast, out);
}
